// BlockV3_84155589198357
// MI455X (gfx1250) — matrix operand order repaired and checked by reading, not compiled, not run on hardware
//
#include <hip/hip_runtime.h>
#include <hip/hip_bf16.h>
#include <math.h>

typedef __attribute__((ext_vector_type(16))) __bf16 v16bf;
typedef __attribute__((ext_vector_type(8)))  __bf16 v8bf;
typedef __attribute__((ext_vector_type(8)))  float  v8f;
typedef unsigned int u32x4 __attribute__((ext_vector_type(4)));
typedef int          i32x4 __attribute__((ext_vector_type(4)));
typedef int          i32x8 __attribute__((ext_vector_type(8)));

static constexpr int CEMB = 768;
static constexpr int NH   = 12;
static constexpr int DHD  = 64;
static constexpr int BN   = 2;
static constexpr int TN   = 2048;
static constexpr int MTOT = BN * TN;   // 4096
static constexpr int HID  = 3072;

__device__ __forceinline__ v16bf cat16(v8bf lo, v8bf hi) {
  return __builtin_shufflevector(lo, hi, 0,1,2,3,4,5,6,7,8,9,10,11,12,13,14,15);
}
__device__ __forceinline__ v8f wmma_bf16(v16bf a, v16bf b, v8f c) {
  return __builtin_amdgcn_wmma_f32_16x16x32_bf16(false, a, false, b, (short)0, c, false, false);
}
__device__ __forceinline__ unsigned lds_off32(const void* p) {
  // generic LDS address: aperture in [63:32], byte offset in [31:0]
  return (unsigned)(size_t)p;
}

// ---- Tensor Data Mover: 2D tile (32 cols x nrows of bf16) global -> LDS ----
// LDS rows padded: every 64B stored, skip 16B => 80B row stride (conflict-free b128 reads).
__device__ __forceinline__ void tdm_load_tile(unsigned lds_byte_off, const __bf16* gsrc,
                                              int K, int nrows) {
#if defined(__HIP_DEVICE_COMPILE__)
  unsigned long long ga = (unsigned long long)(size_t)gsrc;
  u32x4 g0;
  g0[0] = 1u;                                        // count=1, user mode, no gather
  g0[1] = lds_byte_off;                              // lds_addr
  g0[2] = (unsigned)ga;                              // global_addr[31:0]
  g0[3] = ((unsigned)(ga >> 32) & 0x1ffffffu) | (2u << 30);  // addr[56:32] | type=2
  i32x8 g1;
  // data_size=1 (2B), pad_enable, pad_interval=3 (16 DW = 64B), pad_amount=3 (4 DW = 16B)
  g1[0] = (int)((1u << 16) | (1u << 20) | (3u << 22) | (3u << 25));
  g1[1] = (int)(((unsigned)K & 0xffffu) << 16);                       // tensor_dim0 lo16
  g1[2] = (int)((((unsigned)K >> 16) & 0xffffu) |
                (((unsigned)nrows & 0xffffu) << 16));                 // dim0 hi | dim1 lo
  g1[3] = (int)(((((unsigned)nrows) >> 16) & 0xffffu) | (32u << 16)); // dim1 hi | tile_dim0=32
  g1[4] = nrows;                                                      // tile_dim1, tile_dim2=0
  g1[5] = K;                                                          // tensor_dim0_stride lo32
  g1[6] = 0;
  g1[7] = 0;
  i32x4 g2 = {0, 0, 0, 0};
  i32x4 g3 = {0, 0, 0, 0};
#if __clang_major__ >= 23
  i32x8 g4 = {0, 0, 0, 0, 0, 0, 0, 0};
  __builtin_amdgcn_tensor_load_to_lds(g0, g1, g2, g3, g4, 0);
#else
  __builtin_amdgcn_tensor_load_to_lds(g0, g1, g2, g3, 0);
#endif
#else
  (void)lds_byte_off; (void)gsrc; (void)K; (void)nrows;
#endif
}
__device__ __forceinline__ void wait_tensor0() {
#if defined(__HIP_DEVICE_COMPILE__)
  __builtin_amdgcn_s_wait_tensorcnt(0);
#endif
}

// ---------------- f32 -> bf16 convert ----------------
__global__ void cvt_bf16(const float* __restrict__ src, __bf16* __restrict__ dst, int n) {
  int i = blockIdx.x * 256 + threadIdx.x;
  if (i < n) dst[i] = (__bf16)src[i];
}

// ---------------- mask prep: bias[b][t] = allowed ? 0 : -1e30 ----------------
__global__ void mask_prep(const int* __restrict__ pmask, const int* __restrict__ cond_len_p,
                          float* __restrict__ bias) {
  int b = blockIdx.x;
  int t = threadIdx.x;
  __shared__ int ssum[8];
  int s = 0;
  for (int i = t; i < TN; i += 256) s += pmask[b * TN + i];
  #pragma unroll
  for (int m = 16; m >= 1; m >>= 1) s += __shfl_xor(s, m, 32);
  if ((t & 31) == 0) ssum[t >> 5] = s;
  __syncthreads();
  int tot = 0;
  #pragma unroll
  for (int w = 0; w < 8; w++) tot += ssum[w];
  int nb = TN - tot;
  int cl = *cond_len_p;
  for (int i = t; i < TN; i += 256)
    bias[b * TN + i] = (i >= cl || i < nb) ? 0.f : -1e30f;
}

// ---------------- LayerNorm (row of 768) -> bf16 ----------------
__global__ __launch_bounds__(256) void ln_cast(const float* __restrict__ x,
                                               const float* __restrict__ g,
                                               const float* __restrict__ beta,
                                               __bf16* __restrict__ out) {
  int row = blockIdx.x;
  const float* xr = x + (size_t)row * CEMB;
  int t = threadIdx.x;
  float v0 = xr[t], v1 = xr[t + 256], v2 = xr[t + 512];
  float s  = v0 + v1 + v2;
  float s2 = v0 * v0 + v1 * v1 + v2 * v2;
  __shared__ float red[16];
  #pragma unroll
  for (int m = 16; m >= 1; m >>= 1) { s += __shfl_xor(s, m, 32); s2 += __shfl_xor(s2, m, 32); }
  if ((t & 31) == 0) { red[t >> 5] = s; red[8 + (t >> 5)] = s2; }
  __syncthreads();
  float ts = 0.f, ts2 = 0.f;
  #pragma unroll
  for (int w = 0; w < 8; w++) { ts += red[w]; ts2 += red[8 + w]; }
  float mu  = ts * (1.f / CEMB);
  float var = ts2 * (1.f / CEMB) - mu * mu;
  float r   = rsqrtf(var + 1e-5f);
  __bf16* o = out + (size_t)row * CEMB;
  o[t]       = (__bf16)((v0 - mu) * r * g[t]       + beta[t]);
  o[t + 256] = (__bf16)((v1 - mu) * r * g[t + 256] + beta[t + 256]);
  o[t + 512] = (__bf16)((v2 - mu) * r * g[t + 512] + beta[t + 512]);
}

// ---------------- WMMA GEMM, both operands TDM-staged through LDS --------------
// out = A[MxK] . W[NxK]^T + bias.  Workgroup: 8 waves, tile M64 x N256.
// Per k-step: wave0 issues 2 tensor_load_to_lds (A 64x32, W 256x32), double buffered.
// Steady state: DMA does all global traffic; waves only read LDS + WMMA.
// MODE 0: bf16 scatter [B,H,T,Dh]   MODE 1: bf16 scatter [B,H,Dh,T]
// MODE 2: f32 + resid row-major     MODE 3: exact GELU -> bf16 row-major
template <int MODE>
__global__ __launch_bounds__(256) void gemm_wmma(const __bf16* __restrict__ A,
                                                 const __bf16* __restrict__ W,
                                                 const float* __restrict__ bias,
                                                 const float* __restrict__ resid,
                                                 void* __restrict__ out, int K, int N) {
  constexpr int STR = 40;  // padded LDS row stride in bf16 elems (80B)
  __shared__ __align__(16) __bf16 bufA[2][64 * STR];    //  5 KB x2
  __shared__ __align__(16) __bf16 bufB[2][256 * STR];   // 20 KB x2
  const int lane = threadIdx.x & 31;
  const int wave = threadIdx.x >> 5;
  const int l16  = lane & 15;
  const int hf   = lane >> 4;
  const int Nblk = blockIdx.x * 256;
  const int Moff = blockIdx.y * 64 + (wave & 1) * 32;

  const __bf16* atile0 = A + (size_t)(blockIdx.y * 64) * K;
  const __bf16* wtile0 = W + (size_t)Nblk * K;
  if (threadIdx.x < 32) {
    tdm_load_tile(lds_off32(&bufA[0][0]), atile0, K, 64);
    tdm_load_tile(lds_off32(&bufB[0][0]), wtile0, K, 256);
  }

  v8f acc[2][4] = {};
  const int arow = ((wave & 1) * 32 + l16) * STR + hf * 16;
  const int brow = ((wave >> 1) * 64 + l16) * STR + hf * 16;

  for (int k0 = 0; k0 < K; k0 += 32) {
    const int cur = (k0 >> 5) & 1;
    if (threadIdx.x < 32) wait_tensor0();
    __syncthreads();                        // publish DMA'd tiles to all waves
    if (threadIdx.x < 32 && (k0 + 32) < K) {
      tdm_load_tile(lds_off32(&bufA[cur ^ 1][0]), atile0 + k0 + 32, K, 64);
      tdm_load_tile(lds_off32(&bufB[cur ^ 1][0]), wtile0 + k0 + 32, K, 256);
    }

    const __bf16* ab = &bufA[cur][0] + arow;
    v16bf af0 = cat16(*(const v8bf*)(ab),            *(const v8bf*)(ab + 8));
    v16bf af1 = cat16(*(const v8bf*)(ab + 16 * STR), *(const v8bf*)(ab + 16 * STR + 8));
    const __bf16* bb = &bufB[cur][0] + brow;
    v16bf bf0 = cat16(*(const v8bf*)(bb),            *(const v8bf*)(bb + 8));
    v16bf bf1 = cat16(*(const v8bf*)(bb + 16 * STR), *(const v8bf*)(bb + 16 * STR + 8));
    v16bf bf2 = cat16(*(const v8bf*)(bb + 32 * STR), *(const v8bf*)(bb + 32 * STR + 8));
    v16bf bf3 = cat16(*(const v8bf*)(bb + 48 * STR), *(const v8bf*)(bb + 48 * STR + 8));
    acc[0][0] = wmma_bf16(af0, bf0, acc[0][0]);
    acc[0][1] = wmma_bf16(af0, bf1, acc[0][1]);
    acc[0][2] = wmma_bf16(af0, bf2, acc[0][2]);
    acc[0][3] = wmma_bf16(af0, bf3, acc[0][3]);
    acc[1][0] = wmma_bf16(af1, bf0, acc[1][0]);
    acc[1][1] = wmma_bf16(af1, bf1, acc[1][1]);
    acc[1][2] = wmma_bf16(af1, bf2, acc[1][2]);
    acc[1][3] = wmma_bf16(af1, bf3, acc[1][3]);
  }

  #pragma unroll
  for (int mi = 0; mi < 2; mi++) {
    #pragma unroll
    for (int ni = 0; ni < 4; ni++) {
      int col  = Nblk + (wave >> 1) * 64 + ni * 16 + l16;
      float bv = bias[col];
      #pragma unroll
      for (int j = 0; j < 8; j++) {
        int m   = Moff + mi * 16 + hf * 8 + j;
        float v = acc[mi][ni][j] + bv;
        if (MODE == 0) {
          int b = m >> 11, t = m & 2047, h = col >> 6, d = col & 63;
          ((__bf16*)out)[(((size_t)(b * NH + h) * TN + t) << 6) + d] = (__bf16)v;
        } else if (MODE == 1) {
          int b = m >> 11, t = m & 2047, h = col >> 6, d = col & 63;
          ((__bf16*)out)[((size_t)(b * NH + h) * DHD + d) * TN + t] = (__bf16)v;
        } else if (MODE == 2) {
          size_t idx = (size_t)m * N + col;
          ((float*)out)[idx] = resid[idx] + v;
        } else {
          float gel = 0.5f * v * (1.f + erff(v * 0.70710678118f));
          ((__bf16*)out)[(size_t)m * N + col] = (__bf16)gel;
        }
      }
    }
  }
}

// ---------------- flash attention: 4 waves x 16 queries, 64-key tiles ----------------
__global__ __launch_bounds__(128) void attn_flash(const __bf16* __restrict__ q,
                                                  const __bf16* __restrict__ k,
                                                  const __bf16* __restrict__ vT,
                                                  const float* __restrict__ maskbias,
                                                  __bf16* __restrict__ ybf) {
  const int lane = threadIdx.x & 31;
  const int wave = threadIdx.x >> 5;
  const int l16  = lane & 15;
  const int hf   = lane >> 4;
  int idx = blockIdx.x;
  int qt  = idx & (TN / 64 - 1);
  int tmp = idx / (TN / 64);
  int h   = tmp % NH;
  int b   = tmp / NH;
  int qbase = qt * 64 + wave * 16;

  const __bf16* qh = q  + (size_t)(b * NH + h) * TN * DHD;
  const __bf16* kh = k  + (size_t)(b * NH + h) * TN * DHD;
  const __bf16* vh = vT + (size_t)(b * NH + h) * DHD * TN;
  const float*  mb = maskbias + (size_t)b * TN;

  __shared__ __align__(16) __bf16 lds[4][16 * 72];   // per-wave P patch, 144B row stride
  __bf16* pw = &lds[wave][0];

  const __bf16* qr = qh + (size_t)(qbase + l16) * DHD;
  v16bf a0 = cat16(*(const v8bf*)(qr + hf * 16),      *(const v8bf*)(qr + 8 + hf * 16));
  v16bf a1 = cat16(*(const v8bf*)(qr + 32 + hf * 16), *(const v8bf*)(qr + 40 + hf * 16));

  v8f O[4] = {};
  float mrow[8], lpart[8];        // lpart: lane-private partial sums (no per-tile reduce)
  #pragma unroll
  for (int j = 0; j < 8; j++) { mrow[j] = -1e30f; lpart[j] = 0.f; }
  const float scale = 0.125f;     // 1/sqrt(64)

  for (int kt = 0; kt < TN; kt += 64) {
    v8f s0 = {}, s1 = {}, s2 = {}, s3 = {};
    {
      const __bf16* kr0 = kh + (size_t)(kt +      l16) * DHD;
      const __bf16* kr1 = kh + (size_t)(kt + 16 + l16) * DHD;
      const __bf16* kr2 = kh + (size_t)(kt + 32 + l16) * DHD;
      const __bf16* kr3 = kh + (size_t)(kt + 48 + l16) * DHD;
      s0 = wmma_bf16(a0, cat16(*(const v8bf*)(kr0 + hf*16),      *(const v8bf*)(kr0 + hf*16 + 8)),  s0);
      s0 = wmma_bf16(a1, cat16(*(const v8bf*)(kr0 + 32 + hf*16), *(const v8bf*)(kr0 + 40 + hf*16)), s0);
      s1 = wmma_bf16(a0, cat16(*(const v8bf*)(kr1 + hf*16),      *(const v8bf*)(kr1 + hf*16 + 8)),  s1);
      s1 = wmma_bf16(a1, cat16(*(const v8bf*)(kr1 + 32 + hf*16), *(const v8bf*)(kr1 + 40 + hf*16)), s1);
      s2 = wmma_bf16(a0, cat16(*(const v8bf*)(kr2 + hf*16),      *(const v8bf*)(kr2 + hf*16 + 8)),  s2);
      s2 = wmma_bf16(a1, cat16(*(const v8bf*)(kr2 + 32 + hf*16), *(const v8bf*)(kr2 + 40 + hf*16)), s2);
      s3 = wmma_bf16(a0, cat16(*(const v8bf*)(kr3 + hf*16),      *(const v8bf*)(kr3 + hf*16 + 8)),  s3);
      s3 = wmma_bf16(a1, cat16(*(const v8bf*)(kr3 + 32 + hf*16), *(const v8bf*)(kr3 + 40 + hf*16)), s3);
    }
    float bias0 = mb[kt +      l16];
    float bias1 = mb[kt + 16 + l16];
    float bias2 = mb[kt + 32 + l16];
    float bias3 = mb[kt + 48 + l16];

    #pragma unroll
    for (int j = 0; j < 8; j++) {
      float v0 = s0[j] * scale + bias0;
      float v1 = s1[j] * scale + bias1;
      float v2 = s2[j] * scale + bias2;
      float v3 = s3[j] * scale + bias3;
      float mx = fmaxf(fmaxf(v0, v1), fmaxf(v2, v3));
      #pragma unroll
      for (int m = 1; m <= 8; m <<= 1) mx = fmaxf(mx, __shfl_xor(mx, m, 16));
      float mnew = fmaxf(mrow[j], mx);
      float corr = __expf(mrow[j] - mnew);
      float e0 = __expf(v0 - mnew);
      float e1 = __expf(v1 - mnew);
      float e2 = __expf(v2 - mnew);
      float e3 = __expf(v3 - mnew);
      lpart[j] = lpart[j] * corr + ((e0 + e1) + (e2 + e3));
      mrow[j]  = mnew;
      #pragma unroll
      for (int n = 0; n < 4; n++) O[n][j] *= corr;
      int r = (hf * 8 + j) * 72;
      pw[r +      l16] = (__bf16)e0;
      pw[r + 16 + l16] = (__bf16)e1;
      pw[r + 32 + l16] = (__bf16)e2;
      pw[r + 48 + l16] = (__bf16)e3;
    }
    asm volatile("s_wait_dscnt 0" ::: "memory");   // DS in-order per wave
    const __bf16* pr = pw + l16 * 72;
    v16bf pa0 = cat16(*(const v8bf*)(pr + hf * 16),      *(const v8bf*)(pr + 8 + hf * 16));
    v16bf pa1 = cat16(*(const v8bf*)(pr + 32 + hf * 16), *(const v8bf*)(pr + 40 + hf * 16));

    #pragma unroll
    for (int n = 0; n < 4; n++) {
      const __bf16* vr = vh + (size_t)(n * 16 + l16) * TN + kt + hf * 16;
      v16bf bv0 = cat16(*(const v8bf*)(vr),      *(const v8bf*)(vr + 8));
      v16bf bv1 = cat16(*(const v8bf*)(vr + 32), *(const v8bf*)(vr + 40));
      O[n] = wmma_bf16(pa0, bv0, O[n]);
      O[n] = wmma_bf16(pa1, bv1, O[n]);
    }
  }

  // one final cross-lane sum for l, then normalize + scatter
  #pragma unroll
  for (int j = 0; j < 8; j++) {
    float l = lpart[j];
    #pragma unroll
    for (int m = 1; m <= 8; m <<= 1) l += __shfl_xor(l, m, 16);
    lpart[j] = 1.f / l;
  }
  #pragma unroll
  for (int n = 0; n < 4; n++) {
    #pragma unroll
    for (int j = 0; j < 8; j++) {
      int t   = qbase + hf * 8 + j;
      int col = h * DHD + n * 16 + l16;
      ybf[(size_t)(b * TN + t) * CEMB + col] = (__bf16)(O[n][j] * lpart[j]);
    }
  }
}

// ---------------- host side ----------------
extern "C" void kernel_launch(void* const* d_in, const int* in_sizes, int n_in,
                              void* d_out, int out_size, void* d_ws, size_t ws_size,
                              hipStream_t stream) {
  const float* x    = (const float*)d_in[0];
  const int*   cond = (const int*)d_in[1];
  const int*   pm   = (const int*)d_in[2];
  const float* g1   = (const float*)d_in[3];
  const float* bln1 = (const float*)d_in[4];
  const float* g2   = (const float*)d_in[5];
  const float* bln2 = (const float*)d_in[6];
  const float* Wq   = (const float*)d_in[7];
  const float* bq   = (const float*)d_in[8];
  const float* Wk   = (const float*)d_in[9];
  const float* bk   = (const float*)d_in[10];
  const float* Wv   = (const float*)d_in[11];
  const float* bv   = (const float*)d_in[12];
  const float* Wp   = (const float*)d_in[13];
  const float* bp   = (const float*)d_in[14];
  const float* W1   = (const float*)d_in[15];
  const float* b1   = (const float*)d_in[16];
  const float* W2   = (const float*)d_in[17];
  const float* b2   = (const float*)d_in[18];

  char* ws = (char*)d_ws;
  size_t off = 0;
  auto alloc = [&](size_t bytes) -> char* {
    char* p = ws + off;
    off += (bytes + 255) & ~(size_t)255;
    return p;
  };
  __bf16* Wqb = (__bf16*)alloc((size_t)CEMB * CEMB * 2);
  __bf16* Wkb = (__bf16*)alloc((size_t)CEMB * CEMB * 2);
  __bf16* Wvb = (__bf16*)alloc((size_t)CEMB * CEMB * 2);
  __bf16* Wpb = (__bf16*)alloc((size_t)CEMB * CEMB * 2);
  __bf16* W1b = (__bf16*)alloc((size_t)HID * CEMB * 2);
  __bf16* W2b = (__bf16*)alloc((size_t)CEMB * HID * 2);
  __bf16* hbf = (__bf16*)alloc((size_t)MTOT * CEMB * 2);
  __bf16* qbf = (__bf16*)alloc((size_t)MTOT * CEMB * 2);
  __bf16* kbf = (__bf16*)alloc((size_t)MTOT * CEMB * 2);
  __bf16* vTb = (__bf16*)alloc((size_t)MTOT * CEMB * 2);
  float*  mbias = (float*)alloc((size_t)BN * TN * 4);
  __bf16* ybf = (__bf16*)alloc((size_t)MTOT * CEMB * 2);
  float*  xa  = (float*)alloc((size_t)MTOT * CEMB * 4);
  __bf16* h2b = (__bf16*)alloc((size_t)MTOT * CEMB * 2);
  __bf16* ubf = (__bf16*)alloc((size_t)MTOT * HID * 2);

  auto cvt = [&](const float* s, __bf16* d, int n) {
    cvt_bf16<<<(n + 255) / 256, 256, 0, stream>>>(s, d, n);
  };
  cvt(Wq, Wqb, CEMB * CEMB);
  cvt(Wk, Wkb, CEMB * CEMB);
  cvt(Wv, Wvb, CEMB * CEMB);
  cvt(Wp, Wpb, CEMB * CEMB);
  cvt(W1, W1b, HID * CEMB);
  cvt(W2, W2b, CEMB * HID);

  mask_prep<<<BN, 256, 0, stream>>>(pm, cond, mbias);
  ln_cast<<<MTOT, 256, 0, stream>>>(x, g1, bln1, hbf);

  dim3 g768(CEMB / 256, MTOT / 64);
  gemm_wmma<0><<<g768, 256, 0, stream>>>(hbf, Wqb, bq, nullptr, qbf, CEMB, CEMB);
  gemm_wmma<0><<<g768, 256, 0, stream>>>(hbf, Wkb, bk, nullptr, kbf, CEMB, CEMB);
  gemm_wmma<1><<<g768, 256, 0, stream>>>(hbf, Wvb, bv, nullptr, vTb, CEMB, CEMB);

  attn_flash<<<BN * NH * (TN / 64), 128, 0, stream>>>(qbf, kbf, vTb, mbias, ybf);

  gemm_wmma<2><<<g768, 256, 0, stream>>>(ybf, Wpb, bp, x, xa, CEMB, CEMB);
  ln_cast<<<MTOT, 256, 0, stream>>>(xa, g2, bln2, h2b);

  dim3 g3072(HID / 256, MTOT / 64);
  gemm_wmma<3><<<g3072, 256, 0, stream>>>(h2b, W1b, b1, nullptr, ubf, CEMB, HID);
  gemm_wmma<2><<<g768, 256, 0, stream>>>(ubf, W2b, b2, xa, (float*)d_out, HID, CEMB);

  (void)in_sizes; (void)n_in; (void)out_size; (void)ws_size;
}
